// LSTMinc_77781857730585
// MI455X (gfx1250) — matrix operand order repaired and checked by reading, not compiled, not run on hardware
//
#include <hip/hip_runtime.h>
#include <hip/hip_bf16.h>
#include <stdint.h>

// ---------------------------------------------------------------------------
// Persistent bf16-WMMA LSTM rollout for MI455X (gfx1250, wave32).
//   grid = 32 blocks x 512 threads (16 waves), each block owns a 16-row batch
//   tile for all 1024 steps. Wave w owns hidden cols [16w,16w+16) and the
//   matching i/f/g/o gate columns -> pointwise LSTM update is wave-local.
//   W_ihN + W1 resident in LDS (bf16, fragment order); W_hh streamed from L2
//   with register double-buffering so L2 latency overlaps the WMMAs.
// ---------------------------------------------------------------------------

typedef __attribute__((ext_vector_type(16))) __bf16 v16bf;
typedef __attribute__((ext_vector_type(8)))  float  v8f;

#define B_TOT   512
#define T_STEPS 1024
#define NDIM    64
#define HDIM    256
#define G4H     1024
#define MTILE   16
#define NBLOCKS (B_TOT / MTILE)
#define NTHREADS 512

// ---- workspace layout (bytes) ----
#define WS_WHH   0          // bf16 [kt=8][n=1024][ki=32]   (524288 B)
#define WS_W1    524288     // bf16 [kt=8][n=256 ][ki=32]   (131072 B)
#define WS_WIHN  655360     // bf16 [kt=2][n=1024][ki=32]   (131072 B)
#define WS_WX    786432     // f32  [k=3][n=1024]           (12288 B)
#define WS_BIAS  798720     // f32  [1024] = b_ih + b_hh    (4096 B)
#define WS_W2    802816     // f32  [256]                   (1024 B)

// ---- dynamic LDS layout (bytes) ----
#define SM_WIHN  0          // 131072
#define SM_W1    131072     // 131072
#define SM_WX    262144     // 12288
#define SM_BIAS  274432     // 4096
#define SM_B1    278528     // 1024
#define SM_W2    279552     // 1024
#define SM_AH    280576     // bf16 [16][256] h staging     (8192)
#define SM_AHDN  288768     // bf16 [16][256] hdn staging   (8192)
#define SM_AN    296960     // bf16 [16][64]  noise staging (2048)
#define SM_DP    299008     // f32  [16] head outputs       (64)
#define SM_X     299072     // f32  [16][3] x_t             (192)
#define SM_TOTAL 299264

__device__ inline float sigm(float x) { return 1.0f / (1.0f + __expf(-x)); }

__device__ inline v8f splat8(float v) {
  v8f r;
#pragma unroll
  for (int e = 0; e < 8; ++e) r[e] = v;
  return r;
}

// A fragment (16x32 bf16) from row-major [m][k] staging, ld in elements.
// Lanes 0-15: M=lane, K = k0+0..15 ; lanes 16-31: K = k0+16..31.
__device__ inline v16bf load_a_frag(const __bf16* row, int ld, int k0) {
  int lane = threadIdx.x & 31;
  int m = lane & 15;
  int k = k0 + ((lane >> 4) << 4);
  union { uint4 q[2]; v16bf v; } u;
  const char* p = (const char*)(row + m * ld + k);
  u.q[0] = *(const uint4*)p;
  u.q[1] = *(const uint4*)(p + 16);   // +8 elements
  return u.v;
}

// B fragment (32x16 bf16) from fragment-ordered [n][32] storage for one k-tile.
// Lanes 0-15: N=lane, K=0..15 ; lanes 16-31: N=lane-16, K=16..31.
__device__ inline v16bf load_b_frag(const __bf16* fragbase, int nbase) {
  int lane = threadIdx.x & 31;
  int n = nbase + (lane & 15);
  int koff = (lane >> 4) << 4;        // 0 or 16
  union { uint4 q[2]; v16bf v; } u;
  const char* p = (const char*)(fragbase + n * 32 + koff);
  u.q[0] = *(const uint4*)p;
  u.q[1] = *(const uint4*)(p + 16);
  return u.v;
}

// ---------------------------------------------------------------------------
// Prep: bf16-convert + reorder weights into WMMA-fragment-friendly layouts.
// ---------------------------------------------------------------------------
__global__ void lstm_prep_kernel(const float* __restrict__ Wih,
                                 const float* __restrict__ Whh,
                                 const float* __restrict__ bih,
                                 const float* __restrict__ bhh,
                                 const float* __restrict__ W1,
                                 const float* __restrict__ W2,
                                 char* __restrict__ ws) {
  __bf16* wsWhh  = (__bf16*)(ws + WS_WHH);
  __bf16* wsW1   = (__bf16*)(ws + WS_W1);
  __bf16* wsWihN = (__bf16*)(ws + WS_WIHN);
  float*  wsWx   = (float*)(ws + WS_WX);
  float*  wsBias = (float*)(ws + WS_BIAS);
  float*  wsW2   = (float*)(ws + WS_W2);

  int idx = blockIdx.x * blockDim.x + threadIdx.x;
  if (idx < 8 * 1024 * 32) {                       // W_hh^T fragments
    int ki = idx & 31, n = (idx >> 5) & 1023, kt = idx >> 15;
    wsWhh[idx] = (__bf16)Whh[n * HDIM + kt * 32 + ki];
    return;
  }
  idx -= 8 * 1024 * 32;
  if (idx < 8 * 256 * 32) {                        // W1^T fragments
    int ki = idx & 31, n = (idx >> 5) & 255, kt = idx >> 13;
    wsW1[idx] = (__bf16)W1[n * HDIM + kt * 32 + ki];
    return;
  }
  idx -= 8 * 256 * 32;
  if (idx < 2 * 1024 * 32) {                       // W_ih noise cols (3..66)
    int ki = idx & 31, n = (idx >> 5) & 1023, kt = idx >> 15;
    wsWihN[idx] = (__bf16)Wih[n * 67 + 3 + kt * 32 + ki];
    return;
  }
  idx -= 2 * 1024 * 32;
  if (idx < 3 * 1024) {                            // W_ih x-cols (0..2), f32
    int k = idx >> 10, n = idx & 1023;
    wsWx[idx] = Wih[n * 67 + k];
    return;
  }
  idx -= 3 * 1024;
  if (idx < 1024) { wsBias[idx] = bih[idx] + bhh[idx]; return; }
  idx -= 1024;
  if (idx < 256) { wsW2[idx] = W2[idx]; return; }
}

// ---------------------------------------------------------------------------
// Persistent recurrence kernel. 16 waves; wave w owns hidden cols [16w,16w+16).
// ---------------------------------------------------------------------------
__global__ void __launch_bounds__(NTHREADS)
lstm_persist_kernel(const float* __restrict__ noise,
                    const float* __restrict__ gap,
                    const float* __restrict__ b1g,
                    const float* __restrict__ b2g,
                    const char*  __restrict__ ws,
                    float* __restrict__ out) {
  extern __shared__ __align__(16) char smem[];
  __bf16* sWihN = (__bf16*)(smem + SM_WIHN);
  __bf16* sW1   = (__bf16*)(smem + SM_W1);
  float*  sWx   = (float*)(smem + SM_WX);
  float*  sBias = (float*)(smem + SM_BIAS);
  float*  sB1   = (float*)(smem + SM_B1);
  float*  sW2   = (float*)(smem + SM_W2);
  __bf16* sAh   = (__bf16*)(smem + SM_AH);
  __bf16* sAhdn = (__bf16*)(smem + SM_AHDN);
  __bf16* sAn   = (__bf16*)(smem + SM_AN);
  float*  sDp   = (float*)(smem + SM_DP);
  float*  sX    = (float*)(smem + SM_X);

  const __bf16* wsWhh = (const __bf16*)(ws + WS_WHH);

  const int tid  = threadIdx.x;
  const int wave = tid >> 5;              // 0..15
  const int lane = tid & 31;
  const int hi8  = (lane >> 4) << 3;      // accumulator M offset of hi half
  const int nlo  = lane & 15;             // accumulator N within tile
  const int jb   = wave * 16;             // this wave's hidden-column base
  const int b0   = blockIdx.x * MTILE;

  // per-thread noise element coordinates (2 contiguous f32 per thread)
  const int nm = (tid * 2) >> 6;          // batch row within tile
  const int nk = (tid * 2) & 63;          // noise dim

  // ---- fill LDS (uint4 copies), zero h staging ----
  for (int i = tid; i < 131072 / 16; i += NTHREADS)
    ((uint4*)sWihN)[i] = ((const uint4*)(ws + WS_WIHN))[i];
  for (int i = tid; i < 131072 / 16; i += NTHREADS)
    ((uint4*)sW1)[i] = ((const uint4*)(ws + WS_W1))[i];
  for (int i = tid; i < 12288 / 16; i += NTHREADS)
    ((uint4*)sWx)[i] = ((const uint4*)(ws + WS_WX))[i];
  for (int i = tid; i < 4096 / 16; i += NTHREADS)
    ((uint4*)sBias)[i] = ((const uint4*)(ws + WS_BIAS))[i];
  for (int i = tid; i < 1024 / 16; i += NTHREADS)
    ((uint4*)sB1)[i] = ((const uint4*)b1g)[i];
  for (int i = tid; i < 1024 / 16; i += NTHREADS)
    ((uint4*)sW2)[i] = ((const uint4*)(ws + WS_W2))[i];
  for (int i = tid; i < 8192 / 16; i += NTHREADS)
    ((uint4*)sAh)[i] = make_uint4(0u, 0u, 0u, 0u);
  __syncthreads();

  const float b2v = b2g[0];

  v8f cst = {};                           // cell state, wave-local 16 columns

  for (int t = 0; t <= T_STEPS; ++t) {
    // ---- issue per-step global loads early; latency hides under head ----
    float2 nv = make_float2(0.0f, 0.0f);
    if (t < T_STEPS) {
      nv = *(const float2*)(noise +
            ((size_t)(b0 + nm) * T_STEPS + t) * NDIM + nk);
      if (t + 1 < T_STEPS)
        __builtin_prefetch(noise +
            ((size_t)(b0 + nm) * T_STEPS + (t + 1)) * NDIM + nk, 0, 0);
    }
    float gp0 = 0.0f, gp1 = 0.0f;
    if (tid < MTILE) {
      const float* gp = gap + ((size_t)(b0 + tid) * (T_STEPS + 1) + t) * 2;
      gp0 = gp[0]; gp1 = gp[1];
    }

    // ================= head layer 1: hdn = tanh(h @ W1^T + b1) ============
    // two accumulator chains (even/odd k-tiles) to break the serial D->C chain
    v8f hacc0 = splat8(sB1[jb + nlo]);
    v8f hacc1 = {};
#pragma unroll
    for (int kt = 0; kt < 8; kt += 2) {
      v16bf af0 = load_a_frag(sAh, HDIM, kt * 32);
      v16bf bf0 = load_b_frag(sW1 + kt * 256 * 32, jb);
      v16bf af1 = load_a_frag(sAh, HDIM, (kt + 1) * 32);
      v16bf bf1 = load_b_frag(sW1 + (kt + 1) * 256 * 32, jb);
      hacc0 = __builtin_amdgcn_wmma_f32_16x16x32_bf16(
          false, af0, false, bf0, (short)0, hacc0, false, false);
      hacc1 = __builtin_amdgcn_wmma_f32_16x16x32_bf16(
          false, af1, false, bf1, (short)0, hacc1, false, false);
    }
    {
      int n = jb + nlo;
#pragma unroll
      for (int e = 0; e < 8; ++e)
        sAhdn[(e + hi8) * HDIM + n] = (__bf16)tanhf(hacc0[e] + hacc1[e]);
    }
    __syncthreads();

    // ====== head layer 2: dp[m] = tanh(hdn[m] . W2 + b2) * 24 =============
    // wave w reduces row m = w over all 256 elements (8 per lane).
    {
      float p = 0.0f;
#pragma unroll
      for (int k = 0; k < 8; ++k)
        p += (float)sAhdn[wave * HDIM + lane * 8 + k] * sW2[lane * 8 + k];
      p += __shfl_xor(p, 1, 32);
      p += __shfl_xor(p, 2, 32);
      p += __shfl_xor(p, 4, 32);
      p += __shfl_xor(p, 8, 32);
      p += __shfl_xor(p, 16, 32);
      if (lane == 0) sDp[wave] = tanhf(p + b2v) * 24.0f;
    }
    __syncthreads();

    // ================= assemble x_t, emit output row ======================
    if (tid < MTILE) {
      float dp = sDp[tid];
      sX[tid * 3 + 0] = gp0; sX[tid * 3 + 1] = gp1; sX[tid * 3 + 2] = dp;
      float* op = out + ((size_t)(b0 + tid) * (T_STEPS + 1) + t) * 3;
      op[0] = gp0; op[1] = gp1; op[2] = dp;
    }
    if (t == T_STEPS) break;

    // stage this step's noise tile (values already in registers)
    {
      __bf16* d = sAn + nm * NDIM + nk;
      d[0] = (__bf16)nv.x; d[1] = (__bf16)nv.y;
    }
    __syncthreads();

    // ================= gates = x@W_ih^T + h@W_hh^T + bias =================
    v8f acc[4];
#pragma unroll
    for (int g = 0; g < 4; ++g)
      acc[g] = splat8(sBias[g * HDIM + jb + nlo]);

    // recurrent part: B tiles streamed from L2-resident fragment buffer,
    // register double-buffered so L2 latency overlaps the WMMAs.
    {
      v16bf b0r = load_b_frag(wsWhh, 0 * HDIM + jb);
      v16bf b1r = load_b_frag(wsWhh, 1 * HDIM + jb);
      v16bf b2r = load_b_frag(wsWhh, 2 * HDIM + jb);
      v16bf b3r = load_b_frag(wsWhh, 3 * HDIM + jb);
      for (int kt = 0; kt < 8; ++kt) {
        v16bf n0 = b0r, n1 = b1r, n2 = b2r, n3 = b3r;
        if (kt < 7) {
          const __bf16* bb = wsWhh + (size_t)(kt + 1) * G4H * 32;
          n0 = load_b_frag(bb, 0 * HDIM + jb);
          n1 = load_b_frag(bb, 1 * HDIM + jb);
          n2 = load_b_frag(bb, 2 * HDIM + jb);
          n3 = load_b_frag(bb, 3 * HDIM + jb);
        }
        v16bf af = load_a_frag(sAh, HDIM, kt * 32);
        acc[0] = __builtin_amdgcn_wmma_f32_16x16x32_bf16(
            false, af, false, b0r, (short)0, acc[0], false, false);
        acc[1] = __builtin_amdgcn_wmma_f32_16x16x32_bf16(
            false, af, false, b1r, (short)0, acc[1], false, false);
        acc[2] = __builtin_amdgcn_wmma_f32_16x16x32_bf16(
            false, af, false, b2r, (short)0, acc[2], false, false);
        acc[3] = __builtin_amdgcn_wmma_f32_16x16x32_bf16(
            false, af, false, b3r, (short)0, acc[3], false, false);
        b0r = n0; b1r = n1; b2r = n2; b3r = n3;
      }
    }
    // noise part: B tiles from LDS
#pragma unroll
    for (int kt = 0; kt < 2; ++kt) {
      v16bf af = load_a_frag(sAn, NDIM, kt * 32);
#pragma unroll
      for (int g = 0; g < 4; ++g) {
        v16bf bf = load_b_frag(sWihN + kt * G4H * 32, g * HDIM + jb);
        acc[g] = __builtin_amdgcn_wmma_f32_16x16x32_bf16(
            false, af, false, bf, (short)0, acc[g], false, false);
      }
    }
    // K=3 x-part (gap0, gap1, head feedback) as f32 FMAs into accumulators
    float xk0[8], xk1[8], xk2[8];
#pragma unroll
    for (int e = 0; e < 8; ++e) {
      int m = e + hi8;
      xk0[e] = sX[m * 3 + 0]; xk1[e] = sX[m * 3 + 1]; xk2[e] = sX[m * 3 + 2];
    }
#pragma unroll
    for (int g = 0; g < 4; ++g) {
      int n = g * HDIM + jb + nlo;
      float w0 = sWx[n], w1 = sWx[1024 + n], w2 = sWx[2048 + n];
#pragma unroll
      for (int e = 0; e < 8; ++e)
        acc[g][e] += xk0[e] * w0 + xk1[e] * w1 + xk2[e] * w2;
    }

    __syncthreads();   // everyone done reading sAh before it is overwritten

    // ================= pointwise LSTM update (wave-local) =================
    {
      int j = jb + nlo;
#pragma unroll
      for (int e = 0; e < 8; ++e) {
        float iv = sigm(acc[0][e]);
        float fv = sigm(acc[1][e]);
        float gv = tanhf(acc[2][e]);
        float ov = sigm(acc[3][e]);
        float cn = fv * cst[e] + iv * gv;
        cst[e] = cn;
        sAh[(e + hi8) * HDIM + j] = (__bf16)(ov * tanhf(cn));
      }
    }
    __syncthreads();
  }
}

// ---------------------------------------------------------------------------
extern "C" void kernel_launch(void* const* d_in, const int* in_sizes, int n_in,
                              void* d_out, int out_size, void* d_ws, size_t ws_size,
                              hipStream_t stream) {
  (void)in_sizes; (void)n_in; (void)out_size; (void)ws_size;
  const float* noise = (const float*)d_in[0];
  const float* gap   = (const float*)d_in[1];
  const float* Wih   = (const float*)d_in[2];
  const float* Whh   = (const float*)d_in[3];
  const float* bih   = (const float*)d_in[4];
  const float* bhh   = (const float*)d_in[5];
  const float* W1    = (const float*)d_in[6];
  const float* b1    = (const float*)d_in[7];
  const float* W2    = (const float*)d_in[8];
  const float* b2    = (const float*)d_in[9];
  float* out = (float*)d_out;
  char*  ws  = (char*)d_ws;

  // 1) weight reorder / bf16 conversion
  {
    int total = 8 * 1024 * 32 + 8 * 256 * 32 + 2 * 1024 * 32 + 3 * 1024 + 1024 + 256;
    int blocks = (total + 255) / 256;
    lstm_prep_kernel<<<blocks, 256, 0, stream>>>(Wih, Whh, bih, bhh, W1, W2, ws);
  }

  // 2) persistent recurrence (dynamic LDS beyond the static cap)
  hipFuncSetAttribute((const void*)lstm_persist_kernel,
                      hipFuncAttributeMaxDynamicSharedMemorySize, SM_TOTAL);
  lstm_persist_kernel<<<NBLOCKS, NTHREADS, SM_TOTAL, stream>>>(
      noise, gap, b1, b2, ws, out);
}
